// JITCompatibleMultiheadAttention_33157147525507
// MI455X (gfx1250) — matrix operand order repaired and checked by reading, compiled, not run on hardware
//
#include <hip/hip_runtime.h>
#include <hip/hip_bf16.h>

// ---------------------------------------------------------------------------
// MI455X multihead attention, f16 WMMA path (v_wmma_f32_16x16x32_f16),
// f32 accumulation + f32 online softmax. Software-pipelined fragment loads.
//   T=2048, B=2, E=1024, H=16, D=64, SCALING=0.125
// ---------------------------------------------------------------------------

typedef __attribute__((ext_vector_type(16))) _Float16 v16h;
typedef __attribute__((ext_vector_type(4)))  _Float16 v4h;
typedef __attribute__((ext_vector_type(8)))  float    v8f;

#define T_SEQ 2048
#define BSZ   2
#define EMB   1024
#define NHEAD 16
#define HDIM  64
#define MROWS (T_SEQ * BSZ)   // 4096

__device__ __forceinline__ v8f wmma_f16(v16h a, v16h b, v8f c) {
    return __builtin_amdgcn_wmma_f32_16x16x32_f16(
        false, a, false, b, (short)0, c, false, false);
}

// A-fragment (16x32 f16, row-major source, leading dim `ld` halves).
// lanes 0-15: row M=lane, halves[0..7]=K0..7, [8..15]=K16..23
// lanes 16-31: row M=lane-16, halves K8..15 / K24..31
__device__ __forceinline__ v16h load_a16(const _Float16* base, int ld) {
    int lane = threadIdx.x & 31;
    int m = lane & 15;
    int g = lane >> 4;
    const _Float16* p0 = base + (size_t)m * ld + g * 8;
    union { v16h v; uint4 q[2]; } u;
    u.q[0] = *(const uint4*)p0;
    u.q[1] = *(const uint4*)(p0 + 16);
    return u.v;
}

// B-fragment (32x16) with B = S^T for row-major S [N x K]: lane holds column
// n=lane&15 (= row n of S), halves K8g..8g+7 / K16+8g..16+8g+7, g=lane>>4.
__device__ __forceinline__ v16h load_b16(const _Float16* base, int ld) {
    int lane = threadIdx.x & 31;
    int n = lane & 15;
    int g = lane >> 4;
    const _Float16* p0 = base + (size_t)n * ld + g * 8;
    union { v16h v; uint4 q[2]; } u;
    u.q[0] = *(const uint4*)p0;
    u.q[1] = *(const uint4*)(p0 + 16);
    return u.v;
}

// ---------------------------------------------------------------------------
// f32 -> f16 convert (scaled), 4 elements/thread
// ---------------------------------------------------------------------------
__global__ void cvt_f32_f16(const float4* __restrict__ in, v4h* __restrict__ out,
                            float scale, int n4) {
    int i = blockIdx.x * blockDim.x + threadIdx.x;
    int stride = gridDim.x * blockDim.x;
    for (; i < n4; i += stride) {
        float4 x = in[i];
        v4h h;
        h.x = (_Float16)(x.x * scale);
        h.y = (_Float16)(x.y * scale);
        h.z = (_Float16)(x.z * scale);
        h.w = (_Float16)(x.w * scale);
        out[i] = h;
    }
}

// ---------------------------------------------------------------------------
// GEMM: C[m,o] = sum_k A[m,k] * W[o,k]  (+ bias[o]*bscale)
// Per wave: 32(M) x 64(N) tile, K chunks of 32, double-buffered fragments.
// mode 0: Q -> [BH,T,D] f16 | mode 1: K -> [BH,T,D] f16
// mode 2: V^T -> [BH,D,T] f16 | mode 3: f32 [M,N] to outF
// ---------------------------------------------------------------------------
__global__ __launch_bounds__(256) void gemm16_kernel(
    const _Float16* __restrict__ A, const _Float16* __restrict__ W,
    const float* __restrict__ bias, float bscale,
    int M, int N, int Kd, int mode,
    _Float16* __restrict__ outH, float* __restrict__ outF)
{
    int wid  = (blockIdx.x * blockDim.x + threadIdx.x) >> 5;
    int lane = threadIdx.x & 31;
    int ntiles = N >> 6;
    int mt = wid / ntiles;
    int nt = wid % ntiles;
    if (mt * 32 >= M) return;                    // wave-uniform

    const _Float16* Arow0 = A + (size_t)(mt * 32) * Kd;
    const _Float16* Arow1 = Arow0 + (size_t)16 * Kd;
    const _Float16* Wrow  = W + (size_t)(nt * 64) * Kd;

    v8f c0[4] = {}, c1[4] = {};

    // prologue: fragments for k=0
    v16h aA = load_a16(Arow0, Kd);
    v16h aB = load_a16(Arow1, Kd);
    v16h bb[4];
#pragma unroll
    for (int j = 0; j < 4; ++j) bb[j] = load_b16(Wrow + (size_t)(j * 16) * Kd, Kd);

    for (int k0 = 0; k0 < Kd; k0 += 32) {
        // prefetch next chunk (clamped; redundant on last iter) BEFORE consuming
        int kp = k0 + 32 < Kd ? k0 + 32 : k0;
        v16h aA_n = load_a16(Arow0 + kp, Kd);
        v16h aB_n = load_a16(Arow1 + kp, Kd);
        v16h bb_n[4];
#pragma unroll
        for (int j = 0; j < 4; ++j)
            bb_n[j] = load_b16(Wrow + (size_t)(j * 16) * Kd + kp, Kd);

#pragma unroll
        for (int j = 0; j < 4; ++j) {
            c0[j] = wmma_f16(aA, bb[j], c0[j]);
            c1[j] = wmma_f16(aB, bb[j], c1[j]);
        }
        aA = aA_n; aB = aB_n;
#pragma unroll
        for (int j = 0; j < 4; ++j) bb[j] = bb_n[j];
    }

    int n = lane & 15, g = lane >> 4;
#pragma unroll
    for (int i = 0; i < 2; ++i) {
        v8f* c = i ? c1 : c0;
#pragma unroll
        for (int j = 0; j < 4; ++j) {
            int o = nt * 64 + j * 16 + n;
            float bv = bias[o] * bscale;
#pragma unroll
            for (int r = 0; r < 8; ++r) {
                int m = mt * 32 + i * 16 + r + 8 * g;
                float val = c[j][r] + bv;
                if (mode == 3) {
                    outF[(size_t)m * N + o] = val;
                } else {
                    int t = m >> 1, b = m & 1;       // row m = t*BSZ + b
                    int h = o >> 6, d = o & 63;
                    int bh = b * NHEAD + h;
                    if (mode == 2)
                        outH[((size_t)bh * HDIM + d) * T_SEQ + t] = (_Float16)val;
                    else
                        outH[((size_t)bh * T_SEQ + t) * HDIM + d] = (_Float16)val;
                }
            }
        }
    }
}

// ---------------------------------------------------------------------------
// Flash attention: one wave = 16 q-rows of one (b,h); k streamed in 32-tiles.
// V fragments issued at loop top (cover softmax latency); next K tile
// prefetched before the current softmax.
// ---------------------------------------------------------------------------
__global__ __launch_bounds__(128) void attn_kernel(
    const _Float16* __restrict__ Q, const _Float16* __restrict__ Km,
    const _Float16* __restrict__ Vt, _Float16* __restrict__ attnO)
{
    __shared__ __align__(16) _Float16 lds_p[4][16 * 32];

    int wave = threadIdx.x >> 5;
    int lane = threadIdx.x & 31;
    int bh   = blockIdx.x >> 5;
    int qblk = blockIdx.x & 31;
    int q0   = qblk * 64 + wave * 16;

    const _Float16* Qh = Q  + (size_t)bh * T_SEQ * HDIM;
    const _Float16* Kh = Km + (size_t)bh * T_SEQ * HDIM;
    const _Float16* Vh = Vt + (size_t)bh * HDIM * T_SEQ;

    v16h qa0 = load_a16(Qh + (size_t)q0 * HDIM +  0, HDIM);
    v16h qa1 = load_a16(Qh + (size_t)q0 * HDIM + 32, HDIM);

    float mrun[8], lrun[8];
    v8f o[4] = {};
#pragma unroll
    for (int r = 0; r < 8; ++r) { mrun[r] = -3.0e38f; lrun[r] = 0.f; }

    _Float16* lp = lds_p[wave];
    int n = lane & 15, g = lane >> 4;

    // prologue: K fragments for tile 0
    v16h kb[4];
    kb[0] = load_b16(Kh +  0, HDIM);
    kb[1] = load_b16(Kh + 32, HDIM);
    kb[2] = load_b16(Kh + (size_t)16 * HDIM +  0, HDIM);
    kb[3] = load_b16(Kh + (size_t)16 * HDIM + 32, HDIM);

    for (int k0 = 0; k0 < T_SEQ; k0 += 32) {
        // V fragments for this tile: issue early, consumed after softmax
        v16h vb[4];
#pragma unroll
        for (int j = 0; j < 4; ++j)
            vb[j] = load_b16(Vh + (size_t)(j * 16) * T_SEQ + k0, T_SEQ);

        // S = Q * K^T  (16x32 scores as two C fragments)
        v8f s0 = {}, s1 = {};
        s0 = wmma_f16(qa0, kb[0], s0);
        s0 = wmma_f16(qa1, kb[1], s0);
        s1 = wmma_f16(qa0, kb[2], s1);
        s1 = wmma_f16(qa1, kb[3], s1);

        // prefetch next tile's K fragments (clamped on last iter)
        int kp = (k0 + 32 < T_SEQ) ? k0 + 32 : k0;
        v16h kb_n[4];
        kb_n[0] = load_b16(Kh + (size_t)kp * HDIM +  0, HDIM);
        kb_n[1] = load_b16(Kh + (size_t)kp * HDIM + 32, HDIM);
        kb_n[2] = load_b16(Kh + (size_t)(kp + 16) * HDIM +  0, HDIM);
        kb_n[3] = load_b16(Kh + (size_t)(kp + 16) * HDIM + 32, HDIM);

        // online softmax (f32); a row lives across a 16-lane half
        float p0[8], p1[8];
#pragma unroll
        for (int r = 0; r < 8; ++r) {
            float mx = fmaxf(s0[r], s1[r]);
            mx = fmaxf(mx, __shfl_xor(mx, 1, 32));
            mx = fmaxf(mx, __shfl_xor(mx, 2, 32));
            mx = fmaxf(mx, __shfl_xor(mx, 4, 32));
            mx = fmaxf(mx, __shfl_xor(mx, 8, 32));
            float mnew = fmaxf(mrun[r], mx);
            float corr = __expf(mrun[r] - mnew);
            mrun[r] = mnew;
            float e0 = __expf(s0[r] - mnew);
            float e1 = __expf(s1[r] - mnew);
            float sm = e0 + e1;
            sm += __shfl_xor(sm, 1, 32);
            sm += __shfl_xor(sm, 2, 32);
            sm += __shfl_xor(sm, 4, 32);
            sm += __shfl_xor(sm, 8, 32);
            lrun[r] = lrun[r] * corr + sm;
            p0[r] = e0; p1[r] = e1;
            o[0][r] *= corr; o[1][r] *= corr; o[2][r] *= corr; o[3][r] *= corr;
        }

        // P: C-layout -> LDS -> A-layout fragment
#pragma unroll
        for (int r = 0; r < 8; ++r) {
            int row = r + 8 * g;
            lp[row * 32 + n]      = (_Float16)p0[r];
            lp[row * 32 + 16 + n] = (_Float16)p1[r];
        }
        asm volatile("s_wait_dscnt 0" ::: "memory");
        v16h pa = load_a16(lp, 32);

        // O += P * V
#pragma unroll
        for (int j = 0; j < 4; ++j) o[j] = wmma_f16(pa, vb[j], o[j]);

#pragma unroll
        for (int j = 0; j < 4; ++j) kb[j] = kb_n[j];
    }

    // epilogue: divide by row sum, scatter to [T*B, E] f16
    int b = bh >> 4, h = bh & 15;
#pragma unroll
    for (int r = 0; r < 8; ++r) {
        float inv = 1.0f / lrun[r];
        int t = q0 + r + 8 * g;
        size_t row = (size_t)(t * BSZ + b) * EMB;
#pragma unroll
        for (int j = 0; j < 4; ++j)
            attnO[row + h * HDIM + j * 16 + n] = (_Float16)(o[j][r] * inv);
    }
}

// ---------------------------------------------------------------------------
// Launch
// ---------------------------------------------------------------------------
extern "C" void kernel_launch(void* const* d_in, const int* in_sizes, int n_in,
                              void* d_out, int out_size, void* d_ws, size_t ws_size,
                              hipStream_t stream) {
    const float* query = (const float*)d_in[0];
    const float* Wq = (const float*)d_in[1];
    const float* bq = (const float*)d_in[2];
    const float* Wk = (const float*)d_in[3];
    const float* bk = (const float*)d_in[4];
    const float* Wv = (const float*)d_in[5];
    const float* bv = (const float*)d_in[6];
    const float* Wo = (const float*)d_in[7];
    const float* bo = (const float*)d_in[8];

    const float SCALING = 0.125f;   // HDIM^-0.5

    // Workspace layout (halves). Total 48 MB.
    _Float16* ws     = (_Float16*)d_ws;
    _Float16* X16    = ws;                      // 4096*1024
    _Float16* Wq16   = ws + 4194304;            // 1024*1024 each
    _Float16* Wk16   = ws + 5242880;
    _Float16* Wv16   = ws + 6291456;
    _Float16* Wo16   = ws + 7340032;
    _Float16* Qb     = ws + 8388608;            // [32,2048,64]
    _Float16* Kb     = ws + 12582912;           // [32,2048,64]
    _Float16* Vtb    = ws + 16777216;           // [32,64,2048]
    _Float16* attn16 = ws + 20971520;           // [4096,1024]

    // 1) f32 -> f16 (SCALING folded into Wq); vectorized x4
    cvt_f32_f16<<<1024, 256, 0, stream>>>((const float4*)query, (v4h*)X16,
                                          1.0f,    (MROWS * EMB) / 4);
    cvt_f32_f16<<< 256, 256, 0, stream>>>((const float4*)Wq, (v4h*)Wq16,
                                          SCALING, (EMB * EMB) / 4);
    cvt_f32_f16<<< 256, 256, 0, stream>>>((const float4*)Wk, (v4h*)Wk16,
                                          1.0f,    (EMB * EMB) / 4);
    cvt_f32_f16<<< 256, 256, 0, stream>>>((const float4*)Wv, (v4h*)Wv16,
                                          1.0f,    (EMB * EMB) / 4);
    cvt_f32_f16<<< 256, 256, 0, stream>>>((const float4*)Wo, (v4h*)Wo16,
                                          1.0f,    (EMB * EMB) / 4);

    // 2) QKV projections: (4096/32)*(1024/64) = 2048 wave-jobs -> 256 blocks
    gemm16_kernel<<<256, 256, 0, stream>>>(X16, Wq16, bq, SCALING,
                                           MROWS, EMB, EMB, 0, Qb,  nullptr);
    gemm16_kernel<<<256, 256, 0, stream>>>(X16, Wk16, bk, 1.0f,
                                           MROWS, EMB, EMB, 1, Kb,  nullptr);
    gemm16_kernel<<<256, 256, 0, stream>>>(X16, Wv16, bv, 1.0f,
                                           MROWS, EMB, EMB, 2, Vtb, nullptr);

    // 3) attention: 32 (b,h) x 32 q-blocks, 4 waves/block
    attn_kernel<<<1024, 128, 0, stream>>>(Qb, Kb, Vtb, attn16);

    // 4) output projection -> f32 d_out
    gemm16_kernel<<<256, 256, 0, stream>>>(attn16, Wo16, bo, 1.0f,
                                           MROWS, EMB, EMB, 3, nullptr, (float*)d_out);
}
